// MambaBlock_48077863911650
// MI455X (gfx1250) — matrix operand order repaired and checked by reading, not compiled, not run on hardware
//
#include <hip/hip_runtime.h>
#include <math.h>

#define NB       4
#define NL       2048
#define DMODEL   1024
#define DINNER   2048
#define DSTATE   16
#define DTRANK   64
#define DCONV    4
#define WXPAD    128   // W_x padded from 96 -> 128 cols

typedef _Float16 v16h __attribute__((ext_vector_type(16)));
typedef _Float16 v8h  __attribute__((ext_vector_type(8)));
typedef float    v8f  __attribute__((ext_vector_type(8)));

__device__ __forceinline__ float sigmoidf_(float x) { return 1.0f / (1.0f + __expf(-x)); }

// ---------------------------------------------------------------------------
// LayerNorm: one block per row of (B*L, D_MODEL)
// ---------------------------------------------------------------------------
__global__ __launch_bounds__(256) void ln_kernel(const float* __restrict__ x,
                                                 const float* __restrict__ g,
                                                 const float* __restrict__ b,
                                                 float* __restrict__ xn) {
    const int row = blockIdx.x;
    const int tid = threadIdx.x;
    __shared__ float red[256];
    const float* xr = x + (size_t)row * DMODEL;
    float v[4];
    float s = 0.0f;
#pragma unroll
    for (int i = 0; i < 4; ++i) { v[i] = xr[tid + 256 * i]; s += v[i]; }
    red[tid] = s; __syncthreads();
    for (int off = 128; off > 0; off >>= 1) {
        if (tid < off) red[tid] += red[tid + off];
        __syncthreads();
    }
    const float mu = red[0] * (1.0f / DMODEL);
    __syncthreads();
    float vs = 0.0f;
#pragma unroll
    for (int i = 0; i < 4; ++i) { float d = v[i] - mu; vs += d * d; }
    red[tid] = vs; __syncthreads();
    for (int off = 128; off > 0; off >>= 1) {
        if (tid < off) red[tid] += red[tid + off];
        __syncthreads();
    }
    const float rstd = rsqrtf(red[0] * (1.0f / DMODEL) + 1e-5f);
    float* outr = xn + (size_t)row * DMODEL;
#pragma unroll
    for (int i = 0; i < 4; ++i) {
        int c = tid + 256 * i;
        outr[c] = (v[i] - mu) * rstd * g[c] + b[c];
    }
}

// ---------------------------------------------------------------------------
// Pad W_x (2048 x 96) -> W_xp (2048 x 128), zero-filled tail
// ---------------------------------------------------------------------------
__global__ __launch_bounds__(256) void pad_wx_kernel(const float* __restrict__ wx,
                                                     float* __restrict__ wxp) {
    const int idx = blockIdx.x * 256 + threadIdx.x;   // over 2048*128
    const int n = idx & (WXPAD - 1);
    const int k = idx >> 7;
    wxp[idx] = (n < 96) ? wx[k * 96 + n] : 0.0f;
}

// ---------------------------------------------------------------------------
// WMMA GEMM: C[M,N] = A[M,K]*B[K,N] (+Cadd if HASADD). fp32 in/out, f16 LDS
// tiles, fp32 WMMA accumulation.
// Block tile 64x128, 8 waves (2x4), each wave owns a 32x32 sub-tile.
// K-step 64 => 8 x v_wmma_f32_16x16x32_f16 per wave per barrier pair.
// Requires: M%64==0, N%128==0, K%64==0 (guaranteed by padding).
// ---------------------------------------------------------------------------
template <bool HASADD>
__global__ __launch_bounds__(256) void gemm_wmma(const float* __restrict__ A,
                                                 const float* __restrict__ B,
                                                 const float* __restrict__ Cadd,
                                                 float* __restrict__ C,
                                                 int K, int lda, int ldb, int ldc) {
    __shared__ __align__(16) _Float16 As[64 * 64];    // [m][k], 8 KB
    __shared__ __align__(16) _Float16 Bs[128 * 64];   // [n][k] transposed, 16 KB

    const int tid   = threadIdx.x;
    const int wave  = tid >> 5;
    const int lane  = tid & 31;
    const int wr    = wave >> 2;       // 0..1  : 32-row band
    const int wc    = wave & 3;        // 0..3  : 32-col band
    const int lrow  = lane & 15;
    const int khalf = lane >> 4;       // K-split per ISA 7.12.2
    const int m0    = blockIdx.y * 64;
    const int n0    = blockIdx.x * 128;

    v8f acc[2][2] = {};

    // staging roles (no bounds checks: dims padded)
    const int ar = tid >> 2;           // 0..63   A row
    const int ak = (tid & 3) * 16;     // 0,16,32,48
    const int bk = tid >> 3;           // 0..31   B k row (and +32)
    const int bn = (tid & 7) * 16;     // 0..112

    for (int k0 = 0; k0 < K; k0 += 64) {
        // ---- stage A: 64x64, 16 floats per thread ----
        {
            const float* ap = A + (size_t)(m0 + ar) * lda + k0 + ak;
#pragma unroll
            for (int h = 0; h < 2; ++h) {
                const float4 a0 = *(const float4*)(ap + h * 8 + 0);
                const float4 a1 = *(const float4*)(ap + h * 8 + 4);
                v8h hv;
                hv[0] = (_Float16)a0.x; hv[1] = (_Float16)a0.y;
                hv[2] = (_Float16)a0.z; hv[3] = (_Float16)a0.w;
                hv[4] = (_Float16)a1.x; hv[5] = (_Float16)a1.y;
                hv[6] = (_Float16)a1.z; hv[7] = (_Float16)a1.w;
                *(v8h*)&As[ar * 64 + ak + h * 8] = hv;
            }
        }
        // ---- stage B: 64x128 -> transposed [n][k], 32 floats per thread ----
#pragma unroll
        for (int kk = 0; kk < 2; ++kk) {
            const int bkk = bk + kk * 32;
            const float* bp = B + (size_t)(k0 + bkk) * ldb + n0 + bn;
            float bv[16];
            *(float4*)&bv[0]  = *(const float4*)(bp + 0);
            *(float4*)&bv[4]  = *(const float4*)(bp + 4);
            *(float4*)&bv[8]  = *(const float4*)(bp + 8);
            *(float4*)&bv[12] = *(const float4*)(bp + 12);
#pragma unroll
            for (int j = 0; j < 16; ++j)
                Bs[(bn + j) * 64 + bkk] = (_Float16)bv[j];
        }
        __syncthreads();

        // ---- 2 K-substeps x 4 WMMA (ISA 7.12.2 fragment layouts) ----
#pragma unroll
        for (int ks = 0; ks < 2; ++ks) {
            v16h af[2], bf[2];
#pragma unroll
            for (int mt = 0; mt < 2; ++mt) {
                const int r = wr * 32 + mt * 16 + lrow;
                const v8h lo = *(const v8h*)&As[r * 64 + ks * 32 + khalf * 8];
                const v8h hi = *(const v8h*)&As[r * 64 + ks * 32 + khalf * 8 + 16];
#pragma unroll
                for (int i = 0; i < 8; ++i) { af[mt][i] = lo[i]; af[mt][i + 8] = hi[i]; }
            }
#pragma unroll
            for (int nt = 0; nt < 2; ++nt) {
                const int c = wc * 32 + nt * 16 + lrow;
                const v8h lo = *(const v8h*)&Bs[c * 64 + ks * 32 + khalf * 8];
                const v8h hi = *(const v8h*)&Bs[c * 64 + ks * 32 + khalf * 8 + 16];
#pragma unroll
                for (int i = 0; i < 8; ++i) { bf[nt][i] = lo[i]; bf[nt][i + 8] = hi[i]; }
            }
            acc[0][0] = __builtin_amdgcn_wmma_f32_16x16x32_f16(false, af[0], false, bf[0], (short)0, acc[0][0], false, false);
            acc[0][1] = __builtin_amdgcn_wmma_f32_16x16x32_f16(false, af[0], false, bf[1], (short)0, acc[0][1], false, false);
            acc[1][0] = __builtin_amdgcn_wmma_f32_16x16x32_f16(false, af[1], false, bf[0], (short)0, acc[1][0], false, false);
            acc[1][1] = __builtin_amdgcn_wmma_f32_16x16x32_f16(false, af[1], false, bf[1], (short)0, acc[1][1], false, false);
        }
        __syncthreads();
    }

    // ---- branch-free epilogue: pointer-walk rows by ldc ----
#pragma unroll
    for (int mt = 0; mt < 2; ++mt) {
#pragma unroll
        for (int nt = 0; nt < 2; ++nt) {
            const int r0 = m0 + wr * 32 + mt * 16 + khalf * 8;
            const int c  = n0 + wc * 32 + nt * 16 + lrow;
            float* cp = C + (size_t)r0 * ldc + c;
            if (HASADD) {
                const float* rp = Cadd + (size_t)r0 * ldc + c;
#pragma unroll
                for (int i = 0; i < 8; ++i) {
                    *cp = acc[mt][nt][i] + *rp;
                    cp += ldc; rp += ldc;
                }
            } else {
#pragma unroll
                for (int i = 0; i < 8; ++i) {
                    *cp = acc[mt][nt][i];
                    cp += ldc;
                }
            }
        }
    }
}

// ---------------------------------------------------------------------------
// Depthwise causal conv (k=4) + SiLU. u is cols [0, D_INNER) of xz.
// ---------------------------------------------------------------------------
__global__ __launch_bounds__(256) void conv_silu_kernel(const float* __restrict__ xz,
                                                        const float* __restrict__ cw,
                                                        const float* __restrict__ cb,
                                                        float* __restrict__ uc) {
    const int idx = blockIdx.x * 256 + threadIdx.x;   // over B*L*D_INNER
    const int d = idx & (DINNER - 1);
    const int l = (idx >> 11) & (NL - 1);
    const int b = idx >> 22;
    float acc = cb[d];
#pragma unroll
    for (int j = 0; j < DCONV; ++j) {
        const int ls = l - (DCONV - 1) + j;
        if (ls >= 0)
            acc += xz[(size_t)(b * NL + ls) * (2 * DINNER) + d] * cw[d * DCONV + j];
    }
    uc[idx] = acc * sigmoidf_(acc);
}

// ---------------------------------------------------------------------------
// dt = softplus(dt_raw + b_dt), in place
// ---------------------------------------------------------------------------
__global__ __launch_bounds__(256) void dt_softplus_kernel(float* __restrict__ dt,
                                                          const float* __restrict__ bdt) {
    const int idx = blockIdx.x * 256 + threadIdx.x;
    const int d = idx & (DINNER - 1);
    const float v = dt[idx] + bdt[d];
    dt[idx] = (v > 20.0f) ? v : log1pf(__expf(v));
}

// ---------------------------------------------------------------------------
// Selective scan: one lane per (b,d) channel, h[16] in VGPRs.
// B_t/C_t chunks staged global->LDS with CDNA5 async-to-LDS DMA
// (ASYNCcnt-tracked), then consumed over 64 steps.
// Fused epilogue: y = (scan + uc*Dskip) * silu(z)
// ---------------------------------------------------------------------------
__global__ __launch_bounds__(256) void scan_kernel(const float* __restrict__ xz,
                                                   const float* __restrict__ uc,
                                                   const float* __restrict__ dt,
                                                   const float* __restrict__ dbc,
                                                   const float* __restrict__ A_log,
                                                   const float* __restrict__ Dskip,
                                                   float* __restrict__ yg) {
    const int tid = threadIdx.x;
    const int b = blockIdx.x >> 3;
    const int d = ((blockIdx.x & 7) << 8) | tid;

    __shared__ __align__(16) float Bsm[64][DSTATE];
    __shared__ __align__(16) float Csm[64][DSTATE];

    float Av[DSTATE], h[DSTATE];
#pragma unroll
    for (int s = 0; s < DSTATE; ++s) {
        Av[s] = -__expf(A_log[d * DSTATE + s]);
        h[s] = 0.0f;
    }
    const float Dv = Dskip[d];

    for (int tb = 0; tb < NL; tb += 64) {
        // async global->LDS staging of B/C chunk: 64 steps x (16+16) floats
#pragma unroll
        for (int j = 0; j < 8; ++j) {
            const int i = tid + j * 256;        // 0..2047
            const int t = i >> 5;
            const int which = (i >> 4) & 1;
            const int s = i & 15;
            const float* gp =
                dbc + (size_t)(b * NL + tb + t) * WXPAD + DTRANK + which * DSTATE + s;
            float* lp = which ? &Csm[t][s] : &Bsm[t][s];
            const unsigned lds_off = (unsigned)(uintptr_t)lp;  // LDS aperture: low 32 bits
            asm volatile("global_load_async_to_lds_b32 %0, %1, off"
                         :: "v"(lds_off), "v"(gp)
                         : "memory");
        }
        asm volatile("s_wait_asynccnt 0" ::: "memory");
        __syncthreads();

        for (int t = 0; t < 64; ++t) {
            const size_t row = (size_t)(b * NL + tb + t);
            const float dtv = dt[row * DINNER + d];
            const float uv  = uc[row * DINNER + d];
            const float zv  = xz[row * (2 * DINNER) + DINNER + d];
            const float du  = dtv * uv;
            float acc = 0.0f;
#pragma unroll
            for (int s = 0; s < DSTATE; ++s) {
                const float dA = __expf(dtv * Av[s]);
                h[s] = h[s] * dA + du * Bsm[t][s];
                acc += h[s] * Csm[t][s];
            }
            const float y = acc + uv * Dv;
            yg[row * DINNER + d] = y * (zv * sigmoidf_(zv));
        }
        __syncthreads();
    }
}

// ---------------------------------------------------------------------------
// Launch
// ---------------------------------------------------------------------------
extern "C" void kernel_launch(void* const* d_in, const int* in_sizes, int n_in,
                              void* d_out, int out_size, void* d_ws, size_t ws_size,
                              hipStream_t stream) {
    (void)in_sizes; (void)n_in; (void)out_size; (void)ws_size;
    const float* x      = (const float*)d_in[0];
    const float* ln_g   = (const float*)d_in[1];
    const float* ln_b   = (const float*)d_in[2];
    const float* W_in   = (const float*)d_in[3];
    const float* conv_w = (const float*)d_in[4];
    const float* conv_b = (const float*)d_in[5];
    const float* W_x    = (const float*)d_in[6];
    const float* W_dt   = (const float*)d_in[7];
    const float* b_dt   = (const float*)d_in[8];
    const float* A_log  = (const float*)d_in[9];
    const float* Dskip  = (const float*)d_in[10];
    const float* W_out  = (const float*)d_in[11];
    float* out = (float*)d_out;

    char* ws = (char*)d_ws;
    size_t off = 0;
    auto alloc = [&](size_t bytes) -> void* {
        void* p = ws + off;
        off += (bytes + 255) & ~(size_t)255;
        return p;
    };
    const int MROWS = NB * NL; // 8192
    float* xn  = (float*)alloc((size_t)MROWS * DMODEL * 4);        // 32 MB
    float* xz  = (float*)alloc((size_t)MROWS * 2 * DINNER * 4);    // 128 MB
    float* ucb = (float*)alloc((size_t)MROWS * DINNER * 4);        // 64 MB
    float* wxp = (float*)alloc((size_t)DINNER * WXPAD * 4);        // 1 MB
    float* dbc = (float*)alloc((size_t)MROWS * WXPAD * 4);         // 4 MB
    float* dtb = (float*)alloc((size_t)MROWS * DINNER * 4);        // 64 MB
    float* yg  = (float*)alloc((size_t)MROWS * DINNER * 4);        // 64 MB

    const int ntot = MROWS * DINNER; // 16,777,216

    // 0) pad W_x to 128 cols (removes all GEMM tail handling)
    pad_wx_kernel<<<(DINNER * WXPAD) / 256, 256, 0, stream>>>(W_x, wxp);
    // 1) layernorm
    ln_kernel<<<MROWS, 256, 0, stream>>>(x, ln_g, ln_b, xn);
    // 2) xz = xn @ W_in                     M=8192 N=4096 K=1024
    gemm_wmma<false><<<dim3((2 * DINNER) / 128, MROWS / 64), 256, 0, stream>>>(
        xn, W_in, nullptr, xz, DMODEL, DMODEL, 2 * DINNER, 2 * DINNER);
    // 3) uc = silu(conv(u) + conv_b)
    conv_silu_kernel<<<ntot / 256, 256, 0, stream>>>(xz, conv_w, conv_b, ucb);
    // 4) dbc = uc @ W_xp                    M=8192 N=128 K=2048
    gemm_wmma<false><<<dim3(WXPAD / 128, MROWS / 64), 256, 0, stream>>>(
        ucb, wxp, nullptr, dbc, DINNER, DINNER, WXPAD, WXPAD);
    // 5) dt_raw = dt_r @ W_dt               M=8192 N=2048 K=64 (A = cols 0..63 of dbc)
    gemm_wmma<false><<<dim3(DINNER / 128, MROWS / 64), 256, 0, stream>>>(
        dbc, W_dt, nullptr, dtb, DTRANK, WXPAD, DINNER, DINNER);
    // 6) dt = softplus(dt_raw + b_dt)
    dt_softplus_kernel<<<ntot / 256, 256, 0, stream>>>(dtb, b_dt);
    // 7) selective scan + skip + silu(z) gating
    scan_kernel<<<NB * (DINNER / 256), 256, 0, stream>>>(
        xz, ucb, dtb, dbc, A_log, Dskip, yg);
    // 8) out = x + yg @ W_out               M=8192 N=1024 K=2048
    gemm_wmma<true><<<dim3(DMODEL / 128, MROWS / 64), 256, 0, stream>>>(
        yg, W_out, x, out, DINNER, DINNER, DMODEL, DMODEL);
}
